// ImgtoClass_Metric_23656679866492
// MI455X (gfx1250) — 8-bit matrix operand order repaired and checked by reading, compiled, not run on hardware
//
#include <hip/hip_runtime.h>

typedef __attribute__((ext_vector_type(16))) int   v16i;
typedef __attribute__((ext_vector_type(4)))  int   v4i;
typedef __attribute__((ext_vector_type(8)))  float v8f;

#define QN     75
#define CC     640
#define HWN    441
#define WAYN   5
#define SHOTN  5
#define MTILES 28
#define NTILES 138
#define KT8    5
#define NROWS  2205
#define TILE8_BYTES (KT8 * 2048)
#define BTILE_U4    (TILE8_BYTES / 16)

#if defined(__has_builtin)
#if __has_builtin(__builtin_amdgcn_global_load_async_to_lds_b128) && \
    __has_builtin(__builtin_amdgcn_s_wait_asynccnt)
#define USE_ASYNC 1
#endif
#if __has_builtin(__builtin_amdgcn_fmed3f)
#define HAVE_MED3 1
#endif
#endif
#ifndef USE_ASYNC
#define USE_ASYNC 0
#endif
#ifndef HAVE_MED3
#define HAVE_MED3 0
#endif

__device__ __forceinline__ unsigned int f32_to_e4m3(float x) {
    unsigned int u = __float_as_uint(x);
    unsigned int s = (u >> 24) & 0x80u;
    float ax = fminf(fabsf(x), 448.0f);
    unsigned int b = __float_as_uint(ax);
    int exp = (int)(b >> 23);
    unsigned int out;
    if (exp < 121) {
        out = (unsigned int)rintf(ax * 512.0f);
    } else {
        unsigned int m    = b & 0x7fffffu;
        unsigned int keep = m >> 20;
        unsigned int rest = m & 0xfffffu;
        unsigned int rnd  = (rest > 0x80000u) || ((rest == 0x80000u) && (keep & 1u));
        unsigned int mant = keep + rnd;
        int e = exp - 127 + 7;
        if (mant == 8u) { mant = 0u; e += 1; }
        if (e > 15 || (e == 15 && mant == 7u)) { e = 15; mant = 6u; }
        out = ((unsigned int)e << 3) | mant;
    }
    return s | out;
}

__device__ __forceinline__ void top3_insert(float v, float& t0, float& t1, float& t2) {
#if HAVE_MED3
    float nm = fminf(t1, v);
    t2 = fmaxf(t2, nm);
    t1 = __builtin_amdgcn_fmed3f(v, t0, t1);
    t0 = fmaxf(t0, v);
#else
    float m0 = fmaxf(t0, v);
    float l0 = fminf(t0, v);
    float m1 = fmaxf(t1, l0);
    float l1 = fminf(t1, l0);
    float m2 = fmaxf(t2, l1);
    t0 = m0; t1 = m1; t2 = m2;
#endif
}

__device__ __forceinline__ int a_byte_to_k(int t, int hi) {
    int v = t >> 2, p = t & 3;
    int half = v >> 3, vv = v & 7;
    return half * 64 + (vv >> 1) * 16 + (vv & 1) * 4 + p + hi * 8;
}
__device__ __forceinline__ int b_byte_to_k(int t, int hi) {
    int v = t >> 2, p = t & 3;
    return (v >> 2) * 32 + hi * 16 + (v & 3) * 4 + p;
}

__global__ __launch_bounds__(256) void pack_queries8(const float* __restrict__ x1,
                                                     unsigned char* __restrict__ qpack) {
    __shared__ float tile[16 * CC];
    __shared__ float part[256];
    __shared__ float invn[16];
    const int qi = blockIdx.x / MTILES;
    const int mt = blockIdx.x % MTILES;
    const size_t base = (size_t)qi * CC * HWN;

    for (int idx = threadIdx.x; idx < 16 * CC; idx += 256) {
        int r = idx & 15, c = idx >> 4;
        int hw = mt * 16 + r;
        float v = 0.f;
        if (hw < HWN) v = x1[base + (size_t)c * HWN + hw];
        tile[r * CC + c] = v;
    }
    __syncthreads();
    {
        int r = threadIdx.x & 15, i = threadIdx.x >> 4;
        float s = 0.f;
        #pragma unroll
        for (int j = 0; j < CC / 16; ++j) {
            float v = tile[r * CC + i + 16 * j];
            s += v * v;
        }
        part[i * 16 + r] = s;
    }
    __syncthreads();
    if (threadIdx.x < 16) {
        float s = 0.f;
        #pragma unroll
        for (int i = 0; i < 16; ++i) s += part[i * 16 + threadIdx.x];
        invn[threadIdx.x] = (s > 0.f) ? rsqrtf(s) : 0.f;
    }
    __syncthreads();

    const int lane = threadIdx.x & 31;
    const int pr   = threadIdx.x >> 5;
    const int m    = lane & 15, hi = lane >> 4;
    const float sc8 = invn[m] * 8.0f;
    const size_t obase = ((size_t)(qi * MTILES + mt) * KT8) * 2048;
    #pragma unroll
    for (int kt = 0; kt < KT8; ++kt) {
        unsigned int lo = 0, hi32 = 0;
        #pragma unroll
        for (int bb = 0; bb < 8; ++bb) {
            int t = pr * 8 + bb;
            int k = kt * 128 + a_byte_to_k(t, hi);
            unsigned int e = f32_to_e4m3(tile[m * CC + k] * sc8);
            if (bb < 4) lo |= e << (8 * bb); else hi32 |= e << (8 * (bb - 4));
        }
        uint2* op = (uint2*)(qpack + obase + (size_t)kt * 2048 + lane * 64 + pr * 8);
        *op = make_uint2(lo, hi32);
    }
}

__global__ __launch_bounds__(256) void pack_support8(const float* __restrict__ x2,
                                                     unsigned char* __restrict__ spack) {
    __shared__ float tile[16 * CC];
    __shared__ float part[256];
    __shared__ float invn[16];
    __shared__ int   rowbase[16];
    const int w  = blockIdx.x / NTILES;
    const int nt = blockIdx.x % NTILES;

    if (threadIdx.x < 16) {
        int ng = nt * 16 + threadIdx.x;
        int rb = -1;
        if (ng < NROWS) {
            int sidx = ng / HWN, hw = ng % HWN;
            rb = (w * SHOTN + sidx) * CC * HWN + hw;
        }
        rowbase[threadIdx.x] = rb;
    }
    __syncthreads();
    for (int idx = threadIdx.x; idx < 16 * CC; idx += 256) {
        int r = idx & 15, c = idx >> 4;
        int rb = rowbase[r];
        float v = 0.f;
        if (rb >= 0) v = x2[(size_t)rb + (size_t)c * HWN];
        tile[r * CC + c] = v;
    }
    __syncthreads();
    {
        int r = threadIdx.x & 15, i = threadIdx.x >> 4;
        float s = 0.f;
        #pragma unroll
        for (int j = 0; j < CC / 16; ++j) {
            float v = tile[r * CC + i + 16 * j];
            s += v * v;
        }
        part[i * 16 + r] = s;
    }
    __syncthreads();
    if (threadIdx.x < 16) {
        float s = 0.f;
        #pragma unroll
        for (int i = 0; i < 16; ++i) s += part[i * 16 + threadIdx.x];
        invn[threadIdx.x] = (s > 0.f) ? rsqrtf(s) : 0.f;
    }
    __syncthreads();

    const int lane = threadIdx.x & 31;
    const int pr   = threadIdx.x >> 5;
    const int n    = lane & 15, hi = lane >> 4;
    const float sc8 = invn[n] * 8.0f;
    const size_t obase = ((size_t)(w * NTILES + nt) * KT8) * 2048;
    #pragma unroll
    for (int kt = 0; kt < KT8; ++kt) {
        unsigned int lo = 0, hi32 = 0;
        #pragma unroll
        for (int bb = 0; bb < 8; ++bb) {
            int t = pr * 8 + bb;
            int k = kt * 128 + a_byte_to_k(t, hi);
            unsigned int e = f32_to_e4m3(tile[n * CC + k] * sc8);
            if (bb < 4) lo |= e << (8 * bb); else hi32 |= e << (8 * (bb - 4));
        }
        uint2* op = (uint2*)(spack + obase + (size_t)kt * 2048 + lane * 64 + pr * 8);
        *op = make_uint2(lo, hi32);
    }
}

#define COMPUTE_TILE(BUF, MASKED)                                             \
    {                                                                         \
        v8f acc0 = {0.f, 0.f, 0.f, 0.f, 0.f, 0.f, 0.f, 0.f};                  \
        v8f acc1 = {0.f, 0.f, 0.f, 0.f, 0.f, 0.f, 0.f, 0.f};                  \
        const uint4* bb_ = &ldsb[(BUF) * BTILE_U4];                           \
        _Pragma("unroll")                                                     \
        for (int kt_ = 0; kt_ < KT8; ++kt_) {                                 \
            union { v16i v; uint4 q[4]; } b_;                                 \
            _Pragma("unroll")                                                 \
            for (int c_ = 0; c_ < 4; ++c_)                                    \
                b_.q[c_] = bb_[kt_ * 128 + c_ * 32 + lane];                   \
            acc0 = __builtin_amdgcn_wmma_f32_16x16x128_fp8_fp8(               \
                a0[kt_], b_.v, (short)0, acc0, false, false);                 \
            acc1 = __builtin_amdgcn_wmma_f32_16x16x128_fp8_fp8(               \
                a1[kt_], b_.v, (short)0, acc1, false, false);                 \
        }                                                                     \
        _Pragma("unroll")                                                     \
        for (int r_ = 0; r_ < 8; ++r_) {                                      \
            float v0_ = acc0[r_], v1_ = acc1[r_];                             \
            if (MASKED) {                                                     \
                v0_ = colvalid ? v0_ : -1e30f;                                \
                v1_ = colvalid ? v1_ : -1e30f;                                \
            }                                                                 \
            top3_insert(v0_, t0a[r_], t1a[r_], t2a[r_]);                      \
            top3_insert(v1_, t0b[r_], t1b[r_], t2b[r_]);                      \
        }                                                                     \
    }

#if USE_ASYNC
#define STAGE_PAIR(DSTBASE, NTFIRST)                                          \
    {                                                                         \
        const uint4* tp_ = sp4 + (size_t)(NTFIRST) * BTILE_U4;                \
        _Pragma("unroll")                                                     \
        for (int j_ = 0; j_ < 6; ++j_) {                                      \
            int u_ = tid + 224 * j_;                                          \
            if (u_ < 2 * BTILE_U4) {                                          \
                int tl_ = (u_ >= BTILE_U4) ? 1 : 0;                           \
                int uu_ = u_ - tl_ * BTILE_U4;                                \
                int idx_ = ((DSTBASE) + tl_) * BTILE_U4 + (uu_ >> 7) * 128 +  \
                           (uu_ & 3) * 32 + ((uu_ & 127) >> 2);               \
                __builtin_amdgcn_global_load_async_to_lds_b128(               \
                    (v4i*)(tp_ + u_), (v4i*)&ldsb[idx_], 0, 0);               \
            }                                                                 \
        }                                                                     \
    }
#else
#define STAGE_LOAD(RG, NTFIRST)                                               \
    {                                                                         \
        const uint4* tp_ = sp4 + (size_t)(NTFIRST) * BTILE_U4;                \
        _Pragma("unroll")                                                     \
        for (int j_ = 0; j_ < 6; ++j_) {                                      \
            int u_ = tid + 224 * j_;                                          \
            if (u_ < 2 * BTILE_U4) RG[j_] = tp_[u_];                          \
        }                                                                     \
    }
#define STAGE_STORE(RG, DSTBASE)                                              \
    {                                                                         \
        _Pragma("unroll")                                                     \
        for (int j_ = 0; j_ < 6; ++j_) {                                      \
            int u_ = tid + 224 * j_;                                          \
            if (u_ < 2 * BTILE_U4) {                                          \
                int tl_ = (u_ >= BTILE_U4) ? 1 : 0;                           \
                int uu_ = u_ - tl_ * BTILE_U4;                                \
                int idx_ = ((DSTBASE) + tl_) * BTILE_U4 + (uu_ >> 7) * 128 +  \
                           (uu_ & 3) * 32 + ((uu_ & 127) >> 2);               \
                ldsb[idx_] = RG[j_];                                          \
            }                                                                 \
        }                                                                     \
    }
#endif

__global__ __launch_bounds__(224) void dn4_main8(const unsigned char* __restrict__ qpack,
                                                 const unsigned char* __restrict__ spack,
                                                 float* __restrict__ out) {
    __shared__ uint4 ldsb[4 * BTILE_U4];
    __shared__ float redbuf[7];

    const int bx   = blockIdx.x;
    const int qi   = bx / WAYN;
    const int w    = bx % WAYN;
    const int tid  = threadIdx.x;
    const int lane = tid & 31;
    const int wave = tid >> 5;

    const uint4* sp4 = (const uint4*)(spack + (size_t)w * NTILES * TILE8_BYTES);
    const bool colvalid = (lane & 15) < (NROWS - (NTILES - 1) * 16);

    float blocksum = 0.f;

    for (int it = 0; it < 2; ++it) {
        const int pair = wave + 7 * it;
        const int mt0  = pair * 2;
        const int mt1  = mt0 + 1;

        v16i a0[KT8], a1[KT8];
        {
            const unsigned char* ap0 =
                qpack + ((size_t)(qi * MTILES + mt0) * KT8) * 2048 + lane * 64;
            const unsigned char* ap1 =
                qpack + ((size_t)(qi * MTILES + mt1) * KT8) * 2048 + lane * 64;
            #pragma unroll
            for (int kt = 0; kt < KT8; ++kt) {
                a0[kt] = *(const v16i*)(ap0 + kt * 2048);
                a1[kt] = *(const v16i*)(ap1 + kt * 2048);
            }
        }

        float t0a[8], t1a[8], t2a[8], t0b[8], t1b[8], t2b[8];
        #pragma unroll
        for (int r = 0; r < 8; ++r) {
            t0a[r] = t1a[r] = t2a[r] = -1e30f;
            t0b[r] = t1b[r] = t2b[r] = -1e30f;
        }

#if USE_ASYNC
        STAGE_PAIR(0, 0);
        __builtin_amdgcn_s_wait_asynccnt(0);
#else
        {
            uint4 rg[6];
            STAGE_LOAD(rg, 0);
            STAGE_STORE(rg, 0);
        }
#endif
        __syncthreads();

        for (int p = 0; p < (NTILES / 2) - 1; ++p) {
            const int base  = (p & 1) * 2;
            const int nbase = 2 - base;
#if USE_ASYNC
            STAGE_PAIR(nbase, 2 * p + 2);
            COMPUTE_TILE(base, 0)
            COMPUTE_TILE(base + 1, 0)
            __builtin_amdgcn_s_wait_asynccnt(0);
#else
            uint4 rg[6];
            STAGE_LOAD(rg, 2 * p + 2);
            COMPUTE_TILE(base, 0)
            COMPUTE_TILE(base + 1, 0)
            STAGE_STORE(rg, nbase);
#endif
            __syncthreads();
        }
        COMPUTE_TILE(0, 0)
        COMPUTE_TILE(1, 1)

        #pragma unroll
        for (int d = 1; d <= 8; d <<= 1) {
            #pragma unroll
            for (int r = 0; r < 8; ++r) {
                float b0 = __shfl_xor(t0a[r], d, 32);
                float b1 = __shfl_xor(t1a[r], d, 32);
                float b2 = __shfl_xor(t2a[r], d, 32);
                top3_insert(b0, t0a[r], t1a[r], t2a[r]);
                top3_insert(b1, t0a[r], t1a[r], t2a[r]);
                top3_insert(b2, t0a[r], t1a[r], t2a[r]);
                float c0 = __shfl_xor(t0b[r], d, 32);
                float c1 = __shfl_xor(t1b[r], d, 32);
                float c2 = __shfl_xor(t2b[r], d, 32);
                top3_insert(c0, t0b[r], t1b[r], t2b[r]);
                top3_insert(c1, t0b[r], t1b[r], t2b[r]);
                top3_insert(c2, t0b[r], t1b[r], t2b[r]);
            }
        }
        float tsa = 0.f, tsb = 0.f;
        #pragma unroll
        for (int r = 0; r < 8; ++r) {
            tsa += t0a[r] + t1a[r] + t2a[r];
            tsb += t0b[r] + t1b[r] + t2b[r];
        }
        tsa += __shfl_xor(tsa, 16, 32);
        tsb += __shfl_xor(tsb, 16, 32);
        blocksum += tsa + tsb;

        __syncthreads();
    }

    if (lane == 0) redbuf[wave] = blocksum;
    __syncthreads();
    if (tid == 0) {
        float s = 0.f;
        #pragma unroll
        for (int i = 0; i < 7; ++i) s += redbuf[i];
        out[bx] = s * (1.0f / 64.0f);
    }
}

extern "C" void kernel_launch(void* const* d_in, const int* in_sizes, int n_in,
                              void* d_out, int out_size, void* d_ws, size_t ws_size,
                              hipStream_t stream) {
    const float* x1 = (const float*)d_in[0];
    const float* x2 = (const float*)d_in[1];
    float* out = (float*)d_out;

    const size_t qpack_bytes = (size_t)QN * MTILES * KT8 * 2048;
    unsigned char* qpack = (unsigned char*)d_ws;
    unsigned char* spack = (unsigned char*)d_ws + qpack_bytes;

    pack_queries8<<<QN * MTILES, 256, 0, stream>>>(x1, qpack);
    pack_support8<<<WAYN * NTILES, 256, 0, stream>>>(x2, spack);
    dn4_main8<<<QN * WAYN, 224, 0, stream>>>(qpack, spack, out);
}
